// Super_MHAT_16784732193198
// MI455X (gfx1250) — matrix operand order repaired and checked by reading, not compiled, not run on hardware
//
#include <hip/hip_runtime.h>

// ---------------- problem constants ----------------
#define N_B    8
#define T_TOK  4608
#define C_DIM  1024
#define M_ST   9
#define K_WIN  5
#define PAD_W  2
#define T_SUP  512            // T_TOK / M_ST
#define H_HDS  16
#define DK     64
#define NT     (N_B * T_SUP)  // 4096 super-token rows

typedef __attribute__((ext_vector_type(16))) __bf16 v16bf;
typedef __attribute__((ext_vector_type(8)))  float  v8f;

union Frag { v16bf v; unsigned int u[8]; };

__device__ __forceinline__ unsigned short f32_to_bf16(float f) {
  unsigned int u = __float_as_uint(f);
  u = (u + 0x7FFFu + ((u >> 16) & 1u)) >> 16;   // round-to-nearest-even
  return (unsigned short)u;
}

// ---------------- K0: super-token mean pool: st[nt,c] = mean_m x[n, tau*M+m, c]
__global__ __launch_bounds__(256) void meanpool_kernel(const float* __restrict__ x,
                                                       float* __restrict__ st) {
  int i = blockIdx.x * 256 + threadIdx.x;
  if (i >= NT * C_DIM) return;
  int c   = i & (C_DIM - 1);
  int nt  = i >> 10;
  int tau = nt & (T_SUP - 1);
  int n   = nt >> 9;
  const float* base = x + (size_t)n * T_TOK * C_DIM + (size_t)(tau * M_ST) * C_DIM + c;
  float s = 0.f;
#pragma unroll
  for (int m = 0; m < M_ST; ++m) s += base[m * C_DIM];
  st[i] = s * (1.0f / M_ST);
}

// ---------------- K1: logits over K-window + softmax -> assoc[nt,m,k]
// one wave32 per (n,tau,m): 5 dot products of length 1024, wave-shuffle reduced
__global__ __launch_bounds__(256) void assoc_kernel(const float* __restrict__ x,
                                                    const float* __restrict__ st,
                                                    float* __restrict__ assoc) {
  int wid  = (blockIdx.x * 256 + threadIdx.x) >> 5;
  int lane = threadIdx.x & 31;
  if (wid >= NT * M_ST) return;
  int m   = wid % M_ST;
  int nt  = wid / M_ST;
  int tau = nt & (T_SUP - 1);
  int n   = nt >> 9;
  const float* tok = x + ((size_t)nt * M_ST + m) * C_DIM;
  float acc[K_WIN] = {0.f, 0.f, 0.f, 0.f, 0.f};
  for (int c = lane; c < C_DIM; c += 32) {
    float tv = tok[c];
#pragma unroll
    for (int k = 0; k < K_WIN; ++k) {
      int tp = tau + k - PAD_W;
      if (tp >= 0 && tp < T_SUP)
        acc[k] += tv * st[(size_t)(n * T_SUP + tp) * C_DIM + c];
    }
  }
#pragma unroll
  for (int k = 0; k < K_WIN; ++k)
    for (int off = 16; off > 0; off >>= 1)
      acc[k] += __shfl_xor(acc[k], off, 32);
  if (lane == 0) {
    const float scale = 0.03125f;  // C^-0.5 (zero-padded windows keep logit 0)
    float mx = -1e30f;
#pragma unroll
    for (int k = 0; k < K_WIN; ++k) { acc[k] *= scale; mx = fmaxf(mx, acc[k]); }
    float e[K_WIN]; float s = 0.f;
#pragma unroll
    for (int k = 0; k < K_WIN; ++k) { e[k] = __expf(acc[k] - mx); s += e[k]; }
    float inv = 1.0f / s;
#pragma unroll
    for (int k = 0; k < K_WIN; ++k) assoc[(size_t)wid * K_WIN + k] = e[k] * inv;
  }
}

// ---------------- K2: asum[nt] = fold(sum_m assoc)
__global__ __launch_bounds__(256) void asum_kernel(const float* __restrict__ assoc,
                                                   float* __restrict__ asum) {
  int nt = blockIdx.x * 256 + threadIdx.x;
  if (nt >= NT) return;
  int tau = nt & (T_SUP - 1);
  int n   = nt >> 9;
  float s = 0.f;
#pragma unroll
  for (int k = 0; k < K_WIN; ++k) {
    int tp = tau + PAD_W - k;
    if (tp < 0 || tp >= T_SUP) continue;
    const float* ap = assoc + (size_t)(n * T_SUP + tp) * M_ST * K_WIN + k;
#pragma unroll
    for (int m = 0; m < M_ST; ++m) s += ap[m * K_WIN];
  }
  asum[nt] = s;
}

// ---------------- K3: st2[nt,c] = fold(tokens x assoc) / (asum+eps), stored bf16
__global__ __launch_bounds__(256) void st2_kernel(const float* __restrict__ x,
                                                  const float* __restrict__ assoc,
                                                  const float* __restrict__ asum,
                                                  unsigned short* __restrict__ st2b) {
  int b   = blockIdx.x;
  int nt  = b >> 2;
  int c   = (b & 3) * 256 + threadIdx.x;
  int tau = nt & (T_SUP - 1);
  int n   = nt >> 9;
  float acc = 0.f;
#pragma unroll
  for (int k = 0; k < K_WIN; ++k) {
    int tp = tau + PAD_W - k;
    if (tp < 0 || tp >= T_SUP) continue;
    size_t abase = (size_t)(n * T_SUP + tp) * M_ST * K_WIN + k;
    const float* xb = x + (size_t)n * T_TOK * C_DIM + (size_t)(tp * M_ST) * C_DIM + c;
#pragma unroll
    for (int m = 0; m < M_ST; ++m)
      acc += xb[m * C_DIM] * assoc[abase + (size_t)m * K_WIN];
  }
  float inv = 1.0f / (asum[nt] + 1e-12f);
  st2b[(size_t)nt * C_DIM + c] = f32_to_bf16(acc * inv);
}

// ---------------- K4: f32 -> bf16 convert
__global__ __launch_bounds__(256) void cvt_kernel(const float* __restrict__ in,
                                                  unsigned short* __restrict__ out, int n) {
  int i = blockIdx.x * 256 + threadIdx.x;
  if (i < n) out[i] = f32_to_bf16(in[i]);
}

// ---------------- K5: WMMA bf16 GEMM: out[Mr,Nc] = A[Mr,Kd] @ W[Nc,Kd]^T + bias
// 2x2 micro-tile per wave (32x32 output): halves A/B re-reads, 4 WMMA per 4 frag loads
template <bool STORE_BF16>
__global__ __launch_bounds__(256) void gemm_kernel(const unsigned short* __restrict__ A,
                                                   const unsigned short* __restrict__ W,
                                                   const float* __restrict__ bias,
                                                   void* __restrict__ outp,
                                                   int Mr, int Nc, int Kd) {
  int job   = blockIdx.x * 8 + (threadIdx.x >> 5);
  int jcols = Nc >> 5;
  int njobs = (Mr >> 5) * jcols;
  if (job >= njobs) return;
  int jr = job / jcols, jc = job % jcols;
  int lane = threadIdx.x & 31;
  int l15 = lane & 15, khalf = lane >> 4;
  const unsigned short* arow0 = A + (size_t)(jr * 32 + l15) * Kd;
  const unsigned short* arow1 = arow0 + (size_t)16 * Kd;
  const unsigned short* wrow0 = W + (size_t)(jc * 32 + l15) * Kd;
  const unsigned short* wrow1 = wrow0 + (size_t)16 * Kd;
  v8f acc00 = {}, acc01 = {}, acc10 = {}, acc11 = {};
  for (int kk = 0; kk < Kd; kk += 32) {
    Frag af0, af1, bf0, bf1;
#pragma unroll
    for (int v = 0; v < 8; ++v) {
      int ka = kk + (v < 4 ? 2 * v : 16 + 2 * (v - 4)) + khalf * 8;  // A 16x32 layout
      af0.u[v] = *(const unsigned int*)(arow0 + ka);
      af1.u[v] = *(const unsigned int*)(arow1 + ka);
      int kb = kk + (v < 4 ? 2 * v : 16 + 2 * (v - 4)) + khalf * 8;  // B 32x16 layout
      bf0.u[v] = *(const unsigned int*)(wrow0 + kb);
      bf1.u[v] = *(const unsigned int*)(wrow1 + kb);
    }
    acc00 = __builtin_amdgcn_wmma_f32_16x16x32_bf16(false, af0.v, false, bf0.v, (short)0, acc00, false, false);
    acc01 = __builtin_amdgcn_wmma_f32_16x16x32_bf16(false, af0.v, false, bf1.v, (short)0, acc01, false, false);
    acc10 = __builtin_amdgcn_wmma_f32_16x16x32_bf16(false, af1.v, false, bf0.v, (short)0, acc10, false, false);
    acc11 = __builtin_amdgcn_wmma_f32_16x16x32_bf16(false, af1.v, false, bf1.v, (short)0, acc11, false, false);
  }
  float bv0 = bias[jc * 32 + l15];
  float bv1 = bias[jc * 32 + 16 + l15];
#pragma unroll
  for (int i = 0; i < 8; ++i) {
    int r0 = jr * 32 + i + 8 * khalf;
    int r1 = r0 + 16;
    size_t c0 = (size_t)jc * 32 + l15;
    float v00 = acc00[i] + bv0, v01 = acc01[i] + bv1;
    float v10 = acc10[i] + bv0, v11 = acc11[i] + bv1;
    if (STORE_BF16) {
      unsigned short* o = (unsigned short*)outp;
      o[(size_t)r0 * Nc + c0]      = f32_to_bf16(v00);
      o[(size_t)r0 * Nc + c0 + 16] = f32_to_bf16(v01);
      o[(size_t)r1 * Nc + c0]      = f32_to_bf16(v10);
      o[(size_t)r1 * Nc + c0 + 16] = f32_to_bf16(v11);
    } else {
      float* o = (float*)outp;
      o[(size_t)r0 * Nc + c0]      = v00;
      o[(size_t)r0 * Nc + c0 + 16] = v01;
      o[(size_t)r1 * Nc + c0]      = v10;
      o[(size_t)r1 * Nc + c0 + 16] = v11;
    }
  }
}

// ---------------- K6: MHA over super-tokens: one block = (n, h, 16-row q tile)
// 4 waves: WMMA S=QK^T into LDS -> cooperative softmax -> WMMA O = P V
__global__ __launch_bounds__(128) void attn_kernel(const unsigned short* __restrict__ Qb,
                                                   const unsigned short* __restrict__ Kb,
                                                   const unsigned short* __restrict__ Vb,
                                                   float* __restrict__ O) {
  __shared__ float          S[16][512];
  __shared__ unsigned short P[16][512];
  __shared__ float          red[16][8];
  __shared__ float          rowsum[16];
  __shared__ float          rowmax[16];

  int b   = blockIdx.x;
  int qt  = b & 31;
  int h   = (b >> 5) & (H_HDS - 1);
  int n   = b >> 9;
  int w   = threadIdx.x >> 5;
  int lane = threadIdx.x & 31;
  int l15 = lane & 15, khalf = lane >> 4;
  int qr0 = qt * 16;

  // Q fragments (16x64 -> two 16x32 A frags), loaded once
  const unsigned short* qrow = Qb + (size_t)(n * T_SUP + qr0 + l15) * C_DIM + h * DK;
  Frag qa0, qa1;
#pragma unroll
  for (int v = 0; v < 8; ++v) {
    int k0 = (v < 4 ? 2 * v : 16 + 2 * (v - 4)) + khalf * 8;
    qa0.u[v] = *(const unsigned int*)(qrow + k0);
    qa1.u[v] = *(const unsigned int*)(qrow + 32 + k0);
  }

  // phase 1: scores into LDS (each wave owns 8 of 32 key tiles)
  for (int jj = 0; jj < 8; ++jj) {
    int j = w * 8 + jj;
    const unsigned short* krow = Kb + (size_t)(n * T_SUP + j * 16 + l15) * C_DIM + h * DK;
    Frag kb0, kb1;
#pragma unroll
    for (int v = 0; v < 8; ++v) {
      int k0 = (v < 4 ? 2 * v : 16 + 2 * (v - 4)) + khalf * 8;
      kb0.u[v] = *(const unsigned int*)(krow + k0);
      kb1.u[v] = *(const unsigned int*)(krow + 32 + k0);
    }
    v8f acc = {};
    acc = __builtin_amdgcn_wmma_f32_16x16x32_bf16(false, qa0.v, false, kb0.v, (short)0, acc, false, false);
    acc = __builtin_amdgcn_wmma_f32_16x16x32_bf16(false, qa1.v, false, kb1.v, (short)0, acc, false, false);
#pragma unroll
    for (int i = 0; i < 8; ++i)
      S[i + 8 * khalf][j * 16 + l15] = acc[i] * 0.125f;  // 1/sqrt(dk)
  }
  __syncthreads();

  // phase 2: softmax stats (8 threads per row), P = exp(S - max) (bf16, unnormalized)
  {
    int row = threadIdx.x >> 3, seg = threadIdx.x & 7, cb = seg * 64;
    float mx = -1e30f;
    for (int c = 0; c < 64; ++c) mx = fmaxf(mx, S[row][cb + c]);
    red[row][seg] = mx;
    __syncthreads();
    if (threadIdx.x < 16) {
      float m2 = red[threadIdx.x][0];
      for (int s2 = 1; s2 < 8; ++s2) m2 = fmaxf(m2, red[threadIdx.x][s2]);
      rowmax[threadIdx.x] = m2;
    }
    __syncthreads();
    float rm = rowmax[row], sm = 0.f;
    for (int c = 0; c < 64; ++c) {
      float e = __expf(S[row][cb + c] - rm);
      P[row][cb + c] = f32_to_bf16(e);
      sm += e;
    }
    red[row][seg] = sm;
    __syncthreads();
    if (threadIdx.x < 16) {
      float s2 = 0.f;
      for (int q = 0; q < 8; ++q) s2 += red[threadIdx.x][q];
      rowsum[threadIdx.x] = s2;
    }
    __syncthreads();
  }

  // phase 3: O = P V (wave w owns d-columns [16w,16w+16)); fold 1/rowsum at store
  int dcol = w * 16 + l15;
  v8f oacc = {};
  for (int kt = 0; kt < 16; ++kt) {
    int kb = kt * 32;
    Frag pa, vfr;
#pragma unroll
    for (int v = 0; v < 8; ++v) {
      int ka = (v < 4 ? 2 * v : 16 + 2 * (v - 4)) + khalf * 8;
      pa.u[v] = *(const unsigned int*)&P[l15][kb + ka];
      int kr = kb + (v < 4 ? 2 * v : 16 + 2 * (v - 4)) + khalf * 8;
      unsigned int lo = Vb[(size_t)(n * T_SUP + kr)     * C_DIM + h * DK + dcol];
      unsigned int hi = Vb[(size_t)(n * T_SUP + kr + 1) * C_DIM + h * DK + dcol];
      vfr.u[v] = lo | (hi << 16);
    }
    oacc = __builtin_amdgcn_wmma_f32_16x16x32_bf16(false, pa.v, false, vfr.v,
                                                   (short)0, oacc, false, false);
  }
#pragma unroll
  for (int i = 0; i < 8; ++i) {
    int lr = i + 8 * khalf;
    O[(size_t)(n * T_SUP + qr0 + lr) * C_DIM + h * DK + dcol] = oacc[i] / rowsum[lr];
  }
}

// ---------------- K7: distribute super-tokens back: out[n,tau*M+m,c] = sum_k ow*assoc
__global__ __launch_bounds__(256) void distribute_kernel(const float* __restrict__ oproj,
                                                         const float* __restrict__ assoc,
                                                         float* __restrict__ out) {
  size_t i = (size_t)blockIdx.x * 256 + threadIdx.x;
  int c    = (int)(i & (C_DIM - 1));
  int rest = (int)(i >> 10);
  int m    = rest % M_ST;
  int nt   = rest / M_ST;
  int tau  = nt & (T_SUP - 1);
  int n    = nt >> 9;
  float acc = 0.f;
#pragma unroll
  for (int k = 0; k < K_WIN; ++k) {
    int tp = tau + k - PAD_W;
    if (tp < 0 || tp >= T_SUP) continue;
    acc += oproj[(size_t)(n * T_SUP + tp) * C_DIM + c] *
           assoc[((size_t)nt * M_ST + m) * K_WIN + k];
  }
  out[i] = acc;
}

// ---------------- host orchestration ----------------
extern "C" void kernel_launch(void* const* d_in, const int* in_sizes, int n_in,
                              void* d_out, int out_size, void* d_ws, size_t ws_size,
                              hipStream_t stream) {
  (void)in_sizes; (void)n_in; (void)out_size; (void)ws_size;
  const float* x  = (const float*)d_in[0];
  const float* wq = (const float*)d_in[1];
  const float* bq = (const float*)d_in[2];
  const float* wk = (const float*)d_in[3];
  const float* bk = (const float*)d_in[4];
  const float* wv = (const float*)d_in[5];
  const float* bv = (const float*)d_in[6];
  const float* wo = (const float*)d_in[7];
  const float* bo = (const float*)d_in[8];
  float* out = (float*)d_out;

  char* p = (char*)d_ws;
  auto take = [&](size_t bytes) -> char* {
    char* r = p; p += (bytes + 255) & ~(size_t)255; return r;
  };
  float*          st    = (float*)take((size_t)NT * C_DIM * 4);
  float*          assoc = (float*)take((size_t)NT * M_ST * K_WIN * 4);
  float*          asum  = (float*)take((size_t)NT * 4);
  unsigned short* st2b  = (unsigned short*)take((size_t)NT * C_DIM * 2);
  unsigned short* wqb   = (unsigned short*)take((size_t)C_DIM * C_DIM * 2);
  unsigned short* wkb   = (unsigned short*)take((size_t)C_DIM * C_DIM * 2);
  unsigned short* wvb   = (unsigned short*)take((size_t)C_DIM * C_DIM * 2);
  unsigned short* wob   = (unsigned short*)take((size_t)C_DIM * C_DIM * 2);
  unsigned short* Qb    = (unsigned short*)take((size_t)NT * C_DIM * 2);
  unsigned short* Kbm   = (unsigned short*)take((size_t)NT * C_DIM * 2);
  unsigned short* Vb    = (unsigned short*)take((size_t)NT * C_DIM * 2);
  float*          Of    = (float*)take((size_t)NT * C_DIM * 4);
  unsigned short* Ob    = (unsigned short*)take((size_t)NT * C_DIM * 2);
  float*          oproj = (float*)take((size_t)NT * C_DIM * 4);

  meanpool_kernel<<<(NT * C_DIM) / 256, 256, 0, stream>>>(x, st);
  assoc_kernel<<<(NT * M_ST) / 8, 256, 0, stream>>>(x, st, assoc);
  asum_kernel<<<NT / 256, 256, 0, stream>>>(assoc, asum);
  st2_kernel<<<NT * 4, 256, 0, stream>>>(x, assoc, asum, st2b);

  int wn = C_DIM * C_DIM;
  cvt_kernel<<<wn / 256, 256, 0, stream>>>(wq, wqb, wn);
  cvt_kernel<<<wn / 256, 256, 0, stream>>>(wk, wkb, wn);
  cvt_kernel<<<wn / 256, 256, 0, stream>>>(wv, wvb, wn);
  cvt_kernel<<<wn / 256, 256, 0, stream>>>(wo, wob, wn);

  int gemm_blocks = ((NT / 32) * (C_DIM / 32)) / 8;  // 512 blocks, 8 wave-jobs each
  gemm_kernel<true><<<gemm_blocks, 256, 0, stream>>>(st2b, wqb, bq, Qb, NT, C_DIM, C_DIM);
  gemm_kernel<true><<<gemm_blocks, 256, 0, stream>>>(st2b, wkb, bk, Kbm, NT, C_DIM, C_DIM);
  gemm_kernel<true><<<gemm_blocks, 256, 0, stream>>>(st2b, wvb, bv, Vb, NT, C_DIM, C_DIM);

  attn_kernel<<<N_B * H_HDS * (T_SUP / 16), 128, 0, stream>>>(Qb, Kbm, Vb, Of);

  cvt_kernel<<<(NT * C_DIM) / 256, 256, 0, stream>>>(Of, Ob, NT * C_DIM);
  gemm_kernel<false><<<gemm_blocks, 256, 0, stream>>>(Ob, wob, bo, oproj, NT, C_DIM, C_DIM);

  distribute_kernel<<<(N_B * T_TOK * C_DIM) / 256, 256, 0, stream>>>(oproj, assoc, out);
}
